// PackedTernaryLinear_91259465105414
// MI455X (gfx1250) — matrix operand order repaired and checked by reading, not compiled, not run on hardware
//
#include <hip/hip_runtime.h>
#include <hip/hip_bf16.h>

typedef __attribute__((ext_vector_type(16))) _Float16 v16h;
typedef __attribute__((ext_vector_type(8)))  _Float16 v8h;
typedef __attribute__((ext_vector_type(8)))  float    v8f;

#define TOKENS 8192
#define IN_F   4096
#define OUT_F  4096
#define ALPHA  0.05f

#define BM 128
#define BN 128

__device__ __forceinline__ _Float16 dec_tern(int c) {
    // code 0 -> 0.0h, 1 -> +1.0h (0x3C00), 2 -> -1.0h (0xBC00)
    unsigned short u = (c == 1) ? (unsigned short)0x3C00
                     : (c == 2) ? (unsigned short)0xBC00
                     : (unsigned short)0x0000;
    return __builtin_bit_cast(_Float16, u);
}

// ---------------------------------------------------------------------------
// Prepass 1: x fp32 -> f16 (one-shot, memory bound)
// ---------------------------------------------------------------------------
__global__ __launch_bounds__(256)
void cvt_x_f16(const float* __restrict__ x, _Float16* __restrict__ xf) {
    const size_t i = ((size_t)blockIdx.x * 256 + threadIdx.x) * 8;
    float4 a = *(const float4*)(x + i);
    float4 b = *(const float4*)(x + i + 4);
    v8h h;
    h[0] = (_Float16)a.x; h[1] = (_Float16)a.y;
    h[2] = (_Float16)a.z; h[3] = (_Float16)a.w;
    h[4] = (_Float16)b.x; h[5] = (_Float16)b.y;
    h[6] = (_Float16)b.z; h[7] = (_Float16)b.w;
    *(v8h*)(xf + i) = h;
}

// ---------------------------------------------------------------------------
// Prepass 2: decode packed ternary (4 codes / int32 low byte) -> f16 W[O][I]
// ---------------------------------------------------------------------------
__global__ __launch_bounds__(256)
void decode_w_f16(const int* __restrict__ pw, _Float16* __restrict__ wf) {
    const size_t g = (size_t)blockIdx.x * 256 + threadIdx.x;   // 16 codes each
    int4 p = *(const int4*)(pw + g * 4);
    v8h b0, b1;
    const int w0 = p.x, w1 = p.y, w2 = p.z, w3 = p.w;
    b0[0] = dec_tern((w0 >> 0) & 3); b0[1] = dec_tern((w0 >> 2) & 3);
    b0[2] = dec_tern((w0 >> 4) & 3); b0[3] = dec_tern((w0 >> 6) & 3);
    b0[4] = dec_tern((w1 >> 0) & 3); b0[5] = dec_tern((w1 >> 2) & 3);
    b0[6] = dec_tern((w1 >> 4) & 3); b0[7] = dec_tern((w1 >> 6) & 3);
    b1[0] = dec_tern((w2 >> 0) & 3); b1[1] = dec_tern((w2 >> 2) & 3);
    b1[2] = dec_tern((w2 >> 4) & 3); b1[3] = dec_tern((w2 >> 6) & 3);
    b1[4] = dec_tern((w3 >> 0) & 3); b1[5] = dec_tern((w3 >> 2) & 3);
    b1[6] = dec_tern((w3 >> 4) & 3); b1[7] = dec_tern((w3 >> 6) & 3);
    *(v8h*)(wf + g * 16)     = b0;
    *(v8h*)(wf + g * 16 + 8) = b1;
}

// ---------------------------------------------------------------------------
// CDNA5 async global->LDS: INST_OFFSET is applied to BOTH addresses (ISA 8.4),
// so a contiguous 64B/thread copy is 1 base pointer + 4 immediates.
// ---------------------------------------------------------------------------
#define ASYNC_CP16_OFF(lds, gp, OFF)                                          \
    asm volatile("global_load_async_to_lds_b128 %0, %1, off offset:" #OFF     \
                 :: "v"(lds), "v"(gp) : "memory")

__device__ __forceinline__ void wait_async0() {
    asm volatile("s_wait_asynccnt 0x0" ::: "memory");
}

#define BK   64
#define LSTR (BK + 8)   // 72 halves = 144B row stride (multiple of 16B)

__device__ __forceinline__ void stage_tile(unsigned la, unsigned lb,
                                           const _Float16* xg,
                                           const _Float16* wg) {
    ASYNC_CP16_OFF(la, xg, 0);
    ASYNC_CP16_OFF(la, xg, 16);
    ASYNC_CP16_OFF(la, xg, 32);
    ASYNC_CP16_OFF(la, xg, 48);
    ASYNC_CP16_OFF(lb, wg, 0);
    ASYNC_CP16_OFF(lb, wg, 16);
    ASYNC_CP16_OFF(lb, wg, 32);
    ASYNC_CP16_OFF(lb, wg, 48);
}

__device__ __forceinline__ void compute_tile(const _Float16 (*A)[LSTR],
                                             const _Float16 (*B)[LSTR],
                                             v8f (&acc)[2][4],
                                             int waveM, int waveN,
                                             int lh, int kA, int kB) {
#pragma unroll
    for (int ks = 0; ks < BK; ks += 32) {
        v16h afrag[2];
#pragma unroll
        for (int mi = 0; mi < 2; ++mi) {
            const _Float16* ap = &A[waveM + mi * 16 + lh][ks];
            v8h lo = *(const v8h*)(ap + kA);
            v8h hi = *(const v8h*)(ap + kA + 16);
            afrag[mi] = __builtin_shufflevector(lo, hi,
                0, 1, 2, 3, 4, 5, 6, 7, 8, 9, 10, 11, 12, 13, 14, 15);
        }
        v16h bfrag[4];
#pragma unroll
        for (int ni = 0; ni < 4; ++ni) {
            const _Float16* bp = &B[waveN + ni * 16 + lh][ks];
            v8h lo = *(const v8h*)(bp + kB);
            v8h hi = *(const v8h*)(bp + kB + 16);
            bfrag[ni] = __builtin_shufflevector(lo, hi,
                0, 1, 2, 3, 4, 5, 6, 7, 8, 9, 10, 11, 12, 13, 14, 15);
        }
#pragma unroll
        for (int mi = 0; mi < 2; ++mi)
#pragma unroll
            for (int ni = 0; ni < 4; ++ni)
                acc[mi][ni] = __builtin_amdgcn_wmma_f32_16x16x32_f16(
                    false, afrag[mi], false, bfrag[ni],
                    (short)0, acc[mi][ni], false, false);
    }
}

// ---------------------------------------------------------------------------
// Main GEMM on pre-converted f16 operands. Double-buffered LDS, async DMA
// staged one K-stage ahead so global->LDS latency hides behind 16 WMMAs.
// ---------------------------------------------------------------------------
__global__ __launch_bounds__(256)
void gemm_f16_wmma(const _Float16* __restrict__ xf,
                   const _Float16* __restrict__ wf,
                   const float* __restrict__ bias,
                   float* __restrict__ out) {
    __shared__ __align__(16) _Float16 As[2][BM][LSTR];
    __shared__ __align__(16) _Float16 Bs[2][BN][LSTR];

    const int t    = threadIdx.x;
    const int lane = t & 31;
    const int wid  = t >> 5;

    const int blockN = blockIdx.x * BN;
    const int blockM = blockIdx.y * BM;

    const int srow = t >> 1;            // 0..127
    const int scol = (t & 1) << 5;      // 0 or 32 halves (64B half-row)

    const int waveM = (wid & 3) << 5;   // 4 waves in M
    const int waveN = (wid >> 2) << 6;  // 2 waves in N

    const int lh = lane & 15;
    const int kA = (lane >> 4) << 3;    // A-frag chunk start (ISA layout)
    const int kB = (lane >> 4) << 3;    // B-frag chunk start

    const v8f vzero = {0.f, 0.f, 0.f, 0.f, 0.f, 0.f, 0.f, 0.f};
    v8f acc[2][4];
#pragma unroll
    for (int i = 0; i < 2; ++i)
#pragma unroll
        for (int j = 0; j < 4; ++j) acc[i][j] = vzero;

    const _Float16* xg = xf + (size_t)(blockM + srow) * IN_F + scol;
    const _Float16* wg = wf + (size_t)(blockN + srow) * IN_F + scol;
    const unsigned la0 = (unsigned)(unsigned long long)&As[0][srow][scol];
    const unsigned la1 = (unsigned)(unsigned long long)&As[1][srow][scol];
    const unsigned lb0 = (unsigned)(unsigned long long)&Bs[0][srow][scol];
    const unsigned lb1 = (unsigned)(unsigned long long)&Bs[1][srow][scol];

    // prologue: fill buffer 0
    stage_tile(la0, lb0, xg, wg);
    wait_async0();
    __syncthreads();

    for (int k0 = 0; k0 < IN_F; k0 += 2 * BK) {
        // phase 0: stage buf1 @ k0+BK while computing buf0 @ k0
        if (k0 + BK < IN_F)
            stage_tile(la1, lb1, xg + k0 + BK, wg + k0 + BK);
        compute_tile(As[0], Bs[0], acc, waveM, waveN, lh, kA, kB);
        wait_async0();
        __syncthreads();

        // phase 1: stage buf0 @ k0+2BK while computing buf1 @ k0+BK
        if (k0 + 2 * BK < IN_F)
            stage_tile(la0, lb0, xg + k0 + 2 * BK, wg + k0 + 2 * BK);
        compute_tile(As[1], Bs[1], acc, waveM, waveN, lh, kA, kB);
        wait_async0();
        __syncthreads();
    }

    float bv[4];
#pragma unroll
    for (int ni = 0; ni < 4; ++ni)
        bv[ni] = bias[blockN + waveN + ni * 16 + lh];

    const int mrow = (lane >> 4) << 3;
#pragma unroll
    for (int mi = 0; mi < 2; ++mi) {
#pragma unroll
        for (int v = 0; v < 8; ++v) {
            const size_t rowOff =
                (size_t)(blockM + waveM + mi * 16 + mrow + v) * OUT_F;
#pragma unroll
            for (int ni = 0; ni < 4; ++ni) {
                const int gn = blockN + waveN + ni * 16 + lh;
                out[rowOff + gn] = acc[mi][ni][v] * ALPHA + bv[ni];
            }
        }
    }
}

// ---------------------------------------------------------------------------
// Fallback: fused decode+convert+GEMM, used if ws too small for prepasses
// ---------------------------------------------------------------------------
#define FBK 32
#define FSTR 40

__global__ __launch_bounds__(256)
void ternary_gemm_fused(const float* __restrict__ x,
                        const int*   __restrict__ pw,
                        const float* __restrict__ bias,
                        float* __restrict__ out) {
    __shared__ __align__(16) _Float16 As[BM][FSTR];
    __shared__ __align__(16) _Float16 Bs[BN][FSTR];

    const int t    = threadIdx.x;
    const int lane = t & 31;
    const int wid  = t >> 5;
    const int blockN = blockIdx.x * BN;
    const int blockM = blockIdx.y * BM;
    const int srow = t >> 1;
    const int scol = (t & 1) << 4;
    const int waveM = (wid & 3) << 5;
    const int waveN = (wid >> 2) << 6;
    const int lh = lane & 15;
    const int kA = (lane >> 4) << 3;
    const int kB = (lane >> 4) << 3;

    const v8f vzero = {0.f, 0.f, 0.f, 0.f, 0.f, 0.f, 0.f, 0.f};
    v8f acc[2][4];
#pragma unroll
    for (int i = 0; i < 2; ++i)
#pragma unroll
        for (int j = 0; j < 4; ++j) acc[i][j] = vzero;

    const float* xrow = x  + (size_t)(blockM + srow) * IN_F + scol;
    const int*   prow = pw + (size_t)(blockN + srow) * (IN_F / 4) + (scol >> 2);

    for (int k0 = 0; k0 < IN_F; k0 += FBK) {
        const float4* xv = (const float4*)(xrow + k0);
        float4 f0 = xv[0], f1 = xv[1], f2 = xv[2], f3 = xv[3];
        int4 p = *(const int4*)(prow + (k0 >> 2));
        if (k0 + FBK < IN_F) {
            __builtin_prefetch(xrow + k0 + FBK, 0, 3);
            __builtin_prefetch(prow + ((k0 + FBK) >> 2), 0, 3);
        }
        __syncthreads();
        v8h a0, a1;
        a0[0] = (_Float16)f0.x; a0[1] = (_Float16)f0.y;
        a0[2] = (_Float16)f0.z; a0[3] = (_Float16)f0.w;
        a0[4] = (_Float16)f1.x; a0[5] = (_Float16)f1.y;
        a0[6] = (_Float16)f1.z; a0[7] = (_Float16)f1.w;
        a1[0] = (_Float16)f2.x; a1[1] = (_Float16)f2.y;
        a1[2] = (_Float16)f2.z; a1[3] = (_Float16)f2.w;
        a1[4] = (_Float16)f3.x; a1[5] = (_Float16)f3.y;
        a1[6] = (_Float16)f3.z; a1[7] = (_Float16)f3.w;
        *(v8h*)&As[srow][scol]     = a0;
        *(v8h*)&As[srow][scol + 8] = a1;
        v8h b0, b1;
        {
            const int w0 = p.x, w1 = p.y, w2 = p.z, w3 = p.w;
            b0[0] = dec_tern((w0 >> 0) & 3); b0[1] = dec_tern((w0 >> 2) & 3);
            b0[2] = dec_tern((w0 >> 4) & 3); b0[3] = dec_tern((w0 >> 6) & 3);
            b0[4] = dec_tern((w1 >> 0) & 3); b0[5] = dec_tern((w1 >> 2) & 3);
            b0[6] = dec_tern((w1 >> 4) & 3); b0[7] = dec_tern((w1 >> 6) & 3);
            b1[0] = dec_tern((w2 >> 0) & 3); b1[1] = dec_tern((w2 >> 2) & 3);
            b1[2] = dec_tern((w2 >> 4) & 3); b1[3] = dec_tern((w2 >> 6) & 3);
            b1[4] = dec_tern((w3 >> 0) & 3); b1[5] = dec_tern((w3 >> 2) & 3);
            b1[6] = dec_tern((w3 >> 4) & 3); b1[7] = dec_tern((w3 >> 6) & 3);
        }
        *(v8h*)&Bs[srow][scol]     = b0;
        *(v8h*)&Bs[srow][scol + 8] = b1;
        __syncthreads();

        v16h afrag[2];
#pragma unroll
        for (int mi = 0; mi < 2; ++mi) {
            const _Float16* ap = &As[waveM + mi * 16 + lh][0];
            v8h lo = *(const v8h*)(ap + kA);
            v8h hi = *(const v8h*)(ap + kA + 16);
            afrag[mi] = __builtin_shufflevector(lo, hi,
                0, 1, 2, 3, 4, 5, 6, 7, 8, 9, 10, 11, 12, 13, 14, 15);
        }
        v16h bfrag[4];
#pragma unroll
        for (int ni = 0; ni < 4; ++ni) {
            const _Float16* bp = &Bs[waveN + ni * 16 + lh][0];
            v8h lo = *(const v8h*)(bp + kB);
            v8h hi = *(const v8h*)(bp + kB + 16);
            bfrag[ni] = __builtin_shufflevector(lo, hi,
                0, 1, 2, 3, 4, 5, 6, 7, 8, 9, 10, 11, 12, 13, 14, 15);
        }
#pragma unroll
        for (int mi = 0; mi < 2; ++mi)
#pragma unroll
            for (int ni = 0; ni < 4; ++ni)
                acc[mi][ni] = __builtin_amdgcn_wmma_f32_16x16x32_f16(
                    false, afrag[mi], false, bfrag[ni],
                    (short)0, acc[mi][ni], false, false);
    }

    float bv[4];
#pragma unroll
    for (int ni = 0; ni < 4; ++ni)
        bv[ni] = bias[blockN + waveN + ni * 16 + lh];
    const int mrow = (lane >> 4) << 3;
#pragma unroll
    for (int mi = 0; mi < 2; ++mi) {
#pragma unroll
        for (int v = 0; v < 8; ++v) {
            const size_t rowOff =
                (size_t)(blockM + waveM + mi * 16 + mrow + v) * OUT_F;
#pragma unroll
            for (int ni = 0; ni < 4; ++ni) {
                const int gn = blockN + waveN + ni * 16 + lh;
                out[rowOff + gn] = acc[mi][ni][v] * ALPHA + bv[ni];
            }
        }
    }
}

// ---------------------------------------------------------------------------
extern "C" void kernel_launch(void* const* d_in, const int* in_sizes, int n_in,
                              void* d_out, int out_size, void* d_ws, size_t ws_size,
                              hipStream_t stream) {
    const float* x    = (const float*)d_in[0];
    const int*   pw   = (const int*)d_in[1];
    const float* bias = (const float*)d_in[2];
    float*       out  = (float*)d_out;

    const size_t xf_bytes = (size_t)TOKENS * IN_F * sizeof(_Float16);  // 64MB
    const size_t wf_bytes = (size_t)OUT_F  * IN_F * sizeof(_Float16);  // 32MB
    dim3 grid(OUT_F / BN, TOKENS / BM);
    dim3 block(256, 1, 1);

    if (ws_size >= xf_bytes + wf_bytes) {
        _Float16* xf = (_Float16*)d_ws;
        _Float16* wf = (_Float16*)((char*)d_ws + xf_bytes);
        cvt_x_f16<<<(TOKENS * (size_t)IN_F) / (256 * 8), block, 0, stream>>>(x, xf);
        decode_w_f16<<<(OUT_F * (size_t)IN_F) / (256 * 16), block, 0, stream>>>(pw, wf);
        gemm_f16_wmma<<<grid, block, 0, stream>>>(xf, wf, bias, out);
    } else {
        ternary_gemm_fused<<<grid, block, 0, stream>>>(x, pw, bias, out);
    }
}
